// AttentionDecoder_24988119728690
// MI455X (gfx1250) — matrix operand order repaired and checked by reading, not compiled, not run on hardware
//
#include <hip/hip_runtime.h>
#include <math.h>

// Problem constants (from the reference)
#define B_   32
#define T_   512
#define U_   64
#define V_   32000
#define H_   512
#define E_   512
#define HE_  1024      // H + E
#define G3_  1536      // 3*H

typedef __attribute__((ext_vector_type(16))) __bf16 v16bf;
typedef __attribute__((ext_vector_type(8)))  float  v8f;

// ---------------------------------------------------------------------------
// WMMA helpers (CDNA5 wave32, v_wmma_f32_16x16x32_bf16)
// ---------------------------------------------------------------------------
__device__ __forceinline__ v8f wmma_bf16(v16bf a, v16bf b, v8f c) {
  // (neg_a, A, neg_b, B, c_mod, C, reuse_a, reuse_b)
  return __builtin_amdgcn_wmma_f32_16x16x32_bf16(false, a, false, b,
                                                 (short)0, c, false, false);
}

// A-fragment: 16x32 (MxK) bf16 from row-major A (lda elems), K offset kb.
// ISA layout: lane l (m=l&15, h=l>>4): VGPR0-3 hold K=kb+8h.., VGPR4-7 K=kb+16+8h..
__device__ __forceinline__ v16bf load_A_frag(const __bf16* A, int lda, int kb) {
  const int l = threadIdx.x & 31;
  const int m = l & 15;
  const int h = l >> 4;
  const __bf16* p = A + (size_t)m * lda + kb;
  union { uint4 u[2]; v16bf v; } t;
  t.u[0] = *(const uint4*)(p + 8 * h);
  t.u[1] = *(const uint4*)(p + 16 + 8 * h);
  return t.v;
}

// B-fragment for C = A x W^T, W row-major [N,K]: lane l (n=l&15, h=l>>4)
// holds B[k][n] = W[n][k] for k = kb+8h..kb+8h+7, kb+16+8h..kb+16+8h+7.
__device__ __forceinline__ v16bf load_Bt_frag(const __bf16* W, int ldw, int kb) {
  const int l = threadIdx.x & 31;
  const int n = l & 15;
  const int h = l >> 4;
  const __bf16* p = W + (size_t)n * ldw + kb;
  union { uint4 u[2]; v16bf v; } t;
  t.u[0] = *(const uint4*)(p + 8 * h);
  t.u[1] = *(const uint4*)(p + 16 + 8 * h);
  return t.v;
}

// C/D fragment store: VGPR r of lane l (n=l&15, h=l>>4) holds C[8h+r][n].
__device__ __forceinline__ void store_C(float* C, int ldc, v8f acc) {
  const int l = threadIdx.x & 31;
  const int n = l & 15;
  const int h = l >> 4;
#pragma unroll
  for (int r = 0; r < 8; ++r) C[(size_t)(8 * h + r) * ldc + n] = acc[r];
}

// ---------------------------------------------------------------------------
// Generic wave-per-tile GEMM: C[M,N] f32 = A[M,K] bf16 @ W[N,K]^T bf16
// (small recurrent GEMMs: latency-bound, direct L2 feeds are fine)
// ---------------------------------------------------------------------------
__global__ void k_gemm_xwt(const __bf16* __restrict__ A,
                           const __bf16* __restrict__ W,
                           float* __restrict__ C,
                           int M, int N, int K) {
  const int wave = (int)((blockIdx.x * blockDim.x + threadIdx.x) >> 5);
  const int ntN = N >> 4;
  const int mt = wave / ntN;
  const int nt = wave % ntN;
  if (mt >= (M >> 4)) return;  // wave-uniform

  const __bf16* Ab = A + (size_t)(mt * 16) * K;
  const __bf16* Wb = W + (size_t)(nt * 16) * K;
  v8f acc = {};
  for (int kb = 0; kb < K; kb += 32) {
    v16bf a = load_A_frag(Ab, K, kb);
    v16bf b = load_Bt_frag(Wb, K, kb);
    acc = wmma_bf16(a, b, acc);
  }
  store_C(C + (size_t)(mt * 16) * N + nt * 16, N, acc);
}

// ---------------------------------------------------------------------------
// Utility kernels
// ---------------------------------------------------------------------------
__global__ void k_cast_bf16(const float* __restrict__ s, __bf16* __restrict__ d,
                            size_t n) {
  size_t i = (size_t)blockIdx.x * blockDim.x + threadIdx.x;
  if (i < n) d[i] = (__bf16)s[i];
}

__global__ void k_init_h(float* h0, float* h1, __bf16* h0b, __bf16* h1b, int n) {
  int i = blockIdx.x * blockDim.x + threadIdx.x;
  if (i < n) {
    h0[i] = 0.f; h1[i] = 0.f;
    h0b[i] = (__bf16)0.f; h1b[i] = (__bf16)0.f;
  }
}

__global__ void k_embed(const int* __restrict__ dec, const float* __restrict__ emb,
                        float* __restrict__ of, __bf16* __restrict__ ob) {
  const int row = blockIdx.x;               // b*U + u
  const int tok = dec[row];
  const float* src = emb + (size_t)tok * H_;
  for (int h = threadIdx.x; h < H_; h += blockDim.x) {
    float v = src[h];
    of[(size_t)row * H_ + h] = v;
    ob[(size_t)row * H_ + h] = (__bf16)v;
  }
}

// ---------------------------------------------------------------------------
// Attention step: one workgroup (256 threads) per batch row.
// ---------------------------------------------------------------------------
__global__ void k_attn(const float* __restrict__ enc_proj,   // [B,T,H]
                       const float* __restrict__ enc_out,    // [B,T,E]
                       const int*   __restrict__ enc_lens,   // [B]
                       const float* __restrict__ embedded,   // [B,U,H]
                       const __bf16* __restrict__ embedded_b,// [B,U,H]
                       const float* __restrict__ h1,         // [B,H]
                       __bf16* __restrict__ X,               // [B,1024]
                       __bf16* __restrict__ Y,               // [U*B,1024]
                       int t) {
  __shared__ float q[H_];
  __shared__ float sc[T_];
  __shared__ float rbuf[256];

  const int b   = blockIdx.x;
  const int tid = threadIdx.x;
  const int len = enc_lens[b];

  for (int h = tid; h < H_; h += 256)
    q[h] = (t == 0) ? embedded[((size_t)b * U_ + t) * H_ + h]
                    : h1[(size_t)b * H_ + h];
  __syncthreads();

  for (int tt = tid; tt < T_; tt += 256) {
    const float* ep = enc_proj + ((size_t)b * T_ + tt) * H_;
    float s = 0.f;
    for (int h = 0; h < H_; h += 4) {
      float4 e = *(const float4*)(ep + h);
      s += e.x * q[h] + e.y * q[h + 1] + e.z * q[h + 2] + e.w * q[h + 3];
    }
    sc[tt] = (tt < len) ? s : -1e30f;
  }
  __syncthreads();

  float m = fmaxf(sc[tid], sc[tid + 256]);
  rbuf[tid] = m; __syncthreads();
  for (int s = 128; s > 0; s >>= 1) {
    if (tid < s) rbuf[tid] = fmaxf(rbuf[tid], rbuf[tid + s]);
    __syncthreads();
  }
  const float mx = rbuf[0]; __syncthreads();

  float e0 = expf(sc[tid] - mx);
  float e1 = expf(sc[tid + 256] - mx);
  rbuf[tid] = e0 + e1; __syncthreads();
  for (int s = 128; s > 0; s >>= 1) {
    if (tid < s) rbuf[tid] += rbuf[tid + s];
    __syncthreads();
  }
  const float inv = 1.f / rbuf[0]; __syncthreads();
  sc[tid]       = e0 * inv;
  sc[tid + 256] = e1 * inv;
  __syncthreads();

  for (int e = tid; e < E_; e += 256) {
    const float* eo = enc_out + (size_t)b * T_ * E_ + e;
    float acc = 0.f;
    for (int tt = 0; tt < T_; ++tt) acc += sc[tt] * eo[(size_t)tt * E_];
    __bf16 cb = (__bf16)acc;
    X[(size_t)b * HE_ + H_ + e] = cb;
    Y[((size_t)t * B_ + b) * HE_ + H_ + e] = cb;
  }
  for (int h = tid; h < H_; h += 256)
    X[(size_t)b * HE_ + h] = embedded_b[((size_t)b * U_ + t) * H_ + h];
}

// ---------------------------------------------------------------------------
// GRU gate math
// ---------------------------------------------------------------------------
__global__ void k_gru_gates(const float* __restrict__ gi,   // [B,1536]
                            const float* __restrict__ gh,   // [B,1536]
                            const float* __restrict__ b_ih,
                            const float* __restrict__ b_hh,
                            float* __restrict__ hf,         // [B,512] in/out
                            __bf16* __restrict__ hb,        // [B,512] out
                            __bf16* __restrict__ Y,         // [U*B,1024] or null
                            int t) {
  int idx = blockIdx.x * blockDim.x + threadIdx.x;  // b*512 + j
  if (idx >= B_ * H_) return;
  const int b = idx >> 9;
  const int j = idx & 511;
  const size_t g = (size_t)b * G3_;

  float ir = gi[g + j]        + b_ih[j];
  float iz = gi[g + 512 + j]  + b_ih[512 + j];
  float in = gi[g + 1024 + j] + b_ih[1024 + j];
  float hr = gh[g + j]        + b_hh[j];
  float hz = gh[g + 512 + j]  + b_hh[512 + j];
  float hn = gh[g + 1024 + j] + b_hh[1024 + j];

  float r = 1.f / (1.f + expf(-(ir + hr)));
  float z = 1.f / (1.f + expf(-(iz + hz)));
  float n = tanhf(in + r * hn);
  float h = hf[idx];
  float o = (1.f - z) * n + z * h;

  hf[idx] = o;
  hb[idx] = (__bf16)o;
  if (Y) Y[((size_t)t * B_ + b) * HE_ + j] = (__bf16)o;
}

// ---------------------------------------------------------------------------
// Final projection: out[b,u,v] = Y[u*B+b,:] . Wout[v,:] + b_out[v]
//
// Block = 128 threads (4 waves). Block tile M=128, N=64, K chunked by 64.
// Double-buffered LDS (A 128x72, B 64x72 bf16, +8 elem row pad = no bank
// conflicts), filled with GLOBAL_LOAD_ASYNC_TO_LDS_B128 (ASYNCcnt) and
// overlapped with WMMA compute. Each wave: M=32 x N=64 register tile,
// 256 v_wmma_f32_16x16x32_bf16 total.
// Grid: (2048/128, 32000/64) = (16, 500).
// ---------------------------------------------------------------------------
#define FMB 128   // block M rows
#define FNB 64    // block N cols
#define FKC 64    // K chunk
#define FLD 72    // padded LDS row stride (elements)

__device__ __forceinline__ void async_ld16(uint32_t lds_off, const __bf16* gsrc) {
  uint64_t g = (uint64_t)(uintptr_t)gsrc;
  asm volatile("global_load_async_to_lds_b128 %0, %1, off"
               :: "v"(lds_off), "v"(g) : "memory");
}

__global__ void __launch_bounds__(128) k_final(
    const __bf16* __restrict__ Y,     // [2048,1024]
    const __bf16* __restrict__ Wo,    // [32000,1024]
    const float*  __restrict__ bo,    // [32000]
    float* __restrict__ out) {        // [B,U,V]
  __shared__ __bf16 As[2][FMB * FLD];
  __shared__ __bf16 Bs[2][FNB * FLD];

  const int tid   = threadIdx.x;
  const int wid   = tid >> 5;
  const int Mbase = blockIdx.x * FMB;
  const int Nbase = blockIdx.y * FNB;

  // Issue one K-chunk of async global->LDS copies (16B granules).
  auto issue = [&](int kc, int buf) {
    const int k0 = kc * FKC;
    // A: 128 rows x 8 granules = 1024 -> 8 per thread
#pragma unroll
    for (int i = 0; i < 8; ++i) {
      int idx = tid + i * 128;
      int r = idx >> 3, c = idx & 7;
      uint32_t lds = (uint32_t)(uintptr_t)&As[buf][r * FLD + c * 8];
      async_ld16(lds, Y + (size_t)(Mbase + r) * HE_ + k0 + c * 8);
    }
    // B: 64 rows x 8 granules = 512 -> 4 per thread
#pragma unroll
    for (int i = 0; i < 4; ++i) {
      int idx = tid + i * 128;
      int r = idx >> 3, c = idx & 7;
      uint32_t lds = (uint32_t)(uintptr_t)&Bs[buf][r * FLD + c * 8];
      async_ld16(lds, Wo + (size_t)(Nbase + r) * HE_ + k0 + c * 8);
    }
  };

  v8f acc[2][4] = {};

  issue(0, 0);
  asm volatile("s_wait_asynccnt 0x0" ::: "memory");
  __syncthreads();

  const int NC = HE_ / FKC;  // 16 chunks
  for (int kc = 0; kc < NC; ++kc) {
    const int buf = kc & 1;
    if (kc + 1 < NC) issue(kc + 1, buf ^ 1);  // prefetch next chunk

    const __bf16* Ab = &As[buf][(wid * 32) * FLD];
    const __bf16* Bb = &Bs[buf][0];
#pragma unroll
    for (int kb = 0; kb < FKC; kb += 32) {
      v16bf a0 = load_A_frag(Ab, FLD, kb);
      v16bf a1 = load_A_frag(Ab + 16 * FLD, FLD, kb);
#pragma unroll
      for (int j = 0; j < 4; ++j) {
        v16bf bf = load_Bt_frag(Bb + j * 16 * FLD, FLD, kb);
        acc[0][j] = wmma_bf16(a0, bf, acc[0][j]);
        acc[1][j] = wmma_bf16(a1, bf, acc[1][j]);
      }
    }
    asm volatile("s_wait_asynccnt 0x0" ::: "memory");
    __syncthreads();
  }

  // Epilogue: bias + scatter to [B,U,V]
  const int l = threadIdx.x & 31;
  const int n = l & 15;
  const int h = l >> 4;
#pragma unroll
  for (int mt = 0; mt < 2; ++mt) {
#pragma unroll
    for (int j = 0; j < 4; ++j) {
      const int v = Nbase + j * 16 + n;
      const float bias = bo[v];
#pragma unroll
      for (int r = 0; r < 8; ++r) {
        const int m = Mbase + wid * 32 + mt * 16 + 8 * h + r;  // m = u*B + b
        const int u = m >> 5;
        const int b = m & 31;
        out[((size_t)b * U_ + u) * V_ + v] = acc[mt][j][r] + bias;
      }
    }
  }
}

// ---------------------------------------------------------------------------
// Host-side launch sequencing
// ---------------------------------------------------------------------------
extern "C" void kernel_launch(void* const* d_in, const int* in_sizes, int n_in,
                              void* d_out, int out_size, void* d_ws, size_t ws_size,
                              hipStream_t stream) {
  const float* encoder_out  = (const float*)d_in[0];
  const int*   encoder_lens = (const int*)d_in[1];
  const int*   decoder_in   = (const int*)d_in[2];
  const float* emb_table    = (const float*)d_in[3];
  const float* W_attn       = (const float*)d_in[4];
  const float* W_ih0        = (const float*)d_in[5];
  const float* W_hh0        = (const float*)d_in[6];
  const float* b_ih0        = (const float*)d_in[7];
  const float* b_hh0        = (const float*)d_in[8];
  const float* W_ih1        = (const float*)d_in[9];
  const float* W_hh1        = (const float*)d_in[10];
  const float* b_ih1        = (const float*)d_in[11];
  const float* b_hh1        = (const float*)d_in[12];
  const float* W_out        = (const float*)d_in[13];
  const float* b_out        = (const float*)d_in[14];
  float* out = (float*)d_out;

  char* base = (char*)d_ws;
  size_t off = 0;
  auto alloc = [&](size_t bytes) -> void* {
    off = (off + 255) & ~(size_t)255;
    void* p = base + off;
    off += bytes;
    return p;
  };

  __bf16* Wout_b  = (__bf16*)alloc((size_t)V_ * HE_ * 2);
  __bf16* enc_b   = (__bf16*)alloc((size_t)B_ * T_ * E_ * 2);
  __bf16* Wattn_b = (__bf16*)alloc((size_t)H_ * E_ * 2);
  __bf16* Wih0_b  = (__bf16*)alloc((size_t)G3_ * HE_ * 2);
  __bf16* Whh0_b  = (__bf16*)alloc((size_t)G3_ * H_ * 2);
  __bf16* Wih1_b  = (__bf16*)alloc((size_t)G3_ * H_ * 2);
  __bf16* Whh1_b  = (__bf16*)alloc((size_t)G3_ * H_ * 2);
  float*  enc_proj   = (float*)alloc((size_t)B_ * T_ * H_ * 4);
  float*  embedded   = (float*)alloc((size_t)B_ * U_ * H_ * 4);
  __bf16* embedded_b = (__bf16*)alloc((size_t)B_ * U_ * H_ * 2);
  float*  h0f = (float*)alloc((size_t)B_ * H_ * 4);
  float*  h1f = (float*)alloc((size_t)B_ * H_ * 4);
  __bf16* h0b = (__bf16*)alloc((size_t)B_ * H_ * 2);
  __bf16* h1b = (__bf16*)alloc((size_t)B_ * H_ * 2);
  __bf16* X   = (__bf16*)alloc((size_t)B_ * HE_ * 2);
  __bf16* Y   = (__bf16*)alloc((size_t)U_ * B_ * HE_ * 2);
  float*  gi  = (float*)alloc((size_t)B_ * G3_ * 4);
  float*  gh  = (float*)alloc((size_t)B_ * G3_ * 4);
  (void)ws_size; (void)in_sizes; (void)n_in; (void)out_size;

  auto cast = [&](const float* s, __bf16* d, size_t n) {
    int blocks = (int)((n + 255) / 256);
    k_cast_bf16<<<blocks, 256, 0, stream>>>(s, d, n);
  };

  // 1) one-time precompute
  cast(W_out, Wout_b, (size_t)V_ * HE_);
  cast(encoder_out, enc_b, (size_t)B_ * T_ * E_);
  cast(W_attn, Wattn_b, (size_t)H_ * E_);
  cast(W_ih0, Wih0_b, (size_t)G3_ * HE_);
  cast(W_hh0, Whh0_b, (size_t)G3_ * H_);
  cast(W_ih1, Wih1_b, (size_t)G3_ * H_);
  cast(W_hh1, Whh1_b, (size_t)G3_ * H_);

  k_embed<<<B_ * U_, 256, 0, stream>>>(decoder_in, emb_table, embedded, embedded_b);

  {  // enc_proj[b,t,h] = enc[b,t,:] . W_attn[h,:]
    int waves = (B_ * T_ / 16) * (H_ / 16);
    k_gemm_xwt<<<(waves + 3) / 4, 128, 0, stream>>>(enc_b, Wattn_b, enc_proj,
                                                    B_ * T_, H_, E_);
  }

  k_init_h<<<(B_ * H_ + 255) / 256, 256, 0, stream>>>(h0f, h1f, h0b, h1b, B_ * H_);

  // 2) sequential recurrence (logits deferred to one big GEMM)
  const int gw = ((B_ / 16) * (G3_ / 16) + 3) / 4;
  for (int t = 0; t < U_; ++t) {
    k_attn<<<B_, 256, 0, stream>>>(enc_proj, encoder_out, encoder_lens,
                                   embedded, embedded_b, h1f, X, Y, t);
    // GRU cell 0
    k_gemm_xwt<<<gw, 128, 0, stream>>>(X,   Wih0_b, gi, B_, G3_, HE_);
    k_gemm_xwt<<<gw, 128, 0, stream>>>(h0b, Whh0_b, gh, B_, G3_, H_);
    k_gru_gates<<<(B_ * H_ + 255) / 256, 256, 0, stream>>>(
        gi, gh, b_ih0, b_hh0, h0f, h0b, (__bf16*)nullptr, t);
    // GRU cell 1 (gh uses OLD h1; gates update h1 and append Y row)
    k_gemm_xwt<<<gw, 128, 0, stream>>>(h0b, Wih1_b, gi, B_, G3_, H_);
    k_gemm_xwt<<<gw, 128, 0, stream>>>(h1b, Whh1_b, gh, B_, G3_, H_);
    k_gru_gates<<<(B_ * H_ + 255) / 256, 256, 0, stream>>>(
        gi, gh, b_ih1, b_hh1, h1f, h1b, Y, t);
  }

  // 3) one big output projection: [2048,1024] x [1024,32000] + bias
  dim3 fgrid(16, 500);
  k_final<<<fgrid, 128, 0, stream>>>(Y, Wout_b, b_out, out);
}
